// HierarchicalMemory_37014028157389
// MI455X (gfx1250) — compile-verified, matrix operand order checked by reading, not run on hardware
//
#include <hip/hip_runtime.h>
#include <hip/hip_bf16.h>

typedef __attribute__((ext_vector_type(16))) __bf16 v16bf;
typedef __attribute__((ext_vector_type(8)))  float  v8f;

#define HDIM 1024
#define SDIM 128
#define TLEN 2048
#define BATCH 4
#define MROWS (BATCH * TLEN) /* 8192 */

#define BM 128
#define BN 128
#define BKT 64              /* K-tile (halfs) */
#define LDSK 72             /* padded K stride in halfs */

__device__ __forceinline__ unsigned short f32_to_bf16(float f) {
    unsigned int u = __float_as_uint(f);
    unsigned int r = u + 0x7FFFu + ((u >> 16) & 1u);   // round-to-nearest-even
    return (unsigned short)(r >> 16);
}

__device__ __forceinline__ float fast_sigmoid(float v) {
    return __builtin_amdgcn_rcpf(1.0f + __expf(-v));
}

// ---- CDNA5 async global->LDS copy (ASYNCcnt-tracked), see CDNA5 ISA
__device__ __forceinline__ void async_b128(const unsigned short* lds_dst, const unsigned short* gsrc) {
    unsigned lds_off = (unsigned)(size_t)lds_dst;                 // flat->LDS: addr[31:0]
    unsigned long long ga = (unsigned long long)(size_t)gsrc;
    asm volatile("global_load_async_to_lds_b128 %0, %1, off"
                 :: "v"(lds_off), "v"(ga) : "memory");
}
__device__ __forceinline__ void wait_async8() { asm volatile("s_wait_asynccnt 0x8" ::: "memory"); }
__device__ __forceinline__ void wait_async0() { asm volatile("s_wait_asynccnt 0x0" ::: "memory"); }

// ------------------------------------------------ f32 -> bf16 elementwise (for x)
__global__ void k_conv(const float* __restrict__ in, unsigned short* __restrict__ out, int n) {
    for (int i = blockIdx.x * blockDim.x + threadIdx.x; i < n; i += gridDim.x * blockDim.x)
        out[i] = f32_to_bf16(in[i]);
}

// ------------------------------------------------ weights f32 [K][N] -> bf16 transposed [N][K]
__global__ void k_convT(const float* __restrict__ in, unsigned short* __restrict__ out, int K, int N) {
    int total = K * N;
    for (int i = blockIdx.x * blockDim.x + threadIdx.x; i < total; i += gridDim.x * blockDim.x) {
        int n = i / K, k = i - n * K;             // output-coalesced
        out[i] = f32_to_bf16(in[(size_t)k * N + n]);
    }
}

// ------------------------------------------------ WMMA GEMM, bf16 A [M][K] x bf16 WT [N][K]
// MODE: 0 none, 1 relu, 2 gate/B combined (sigmoid+bias on first SDIM cols), 3 gate-scale store, 4 gate-scale accumulate
// OUT:  0 f32 -> Cf, 1 bf16 -> Ch, 2 both
template<int MODE, int OUT>
__global__ __launch_bounds__(256) void k_gemm(
    const unsigned short* __restrict__ A, const unsigned short* __restrict__ WT,
    const float* __restrict__ bias, float* __restrict__ Cf, unsigned short* __restrict__ Ch,
    int M, int N, int K, const float* __restrict__ gates, int lvl)
{
    __shared__ unsigned short As[2][BM * LDSK];
    __shared__ unsigned short Bs[2][BN * LDSK];

    const int tid     = threadIdx.x;
    const int lane    = tid & 31;
    const int wave    = tid >> 5;
    const int waveRow = wave >> 2;             // 0..1  -> 64 rows
    const int waveCol = wave & 3;              // 0..3  -> 32 cols
    const int m0 = blockIdx.y * BM;
    const int n0 = blockIdx.x * BN;
    const int halfId = lane >> 4;
    const int l16    = lane & 15;

    v8f acc[4][2];
    const v8f zero = {0.f, 0.f, 0.f, 0.f, 0.f, 0.f, 0.f, 0.f};
#pragma unroll
    for (int i = 0; i < 4; i++)
#pragma unroll
        for (int j = 0; j < 2; j++) acc[i][j] = zero;

    // per-thread slot: 4 b128 for A, 4 b128 for B per tile
    const int srow = tid >> 3;                 // 0..31? no: tid..; slots below
    (void)srow;

    auto stage = [&](int k0, int buf) {
#pragma unroll
        for (int i = 0; i < 4; i++) {
            int s = tid + i * 256;
            int row = s >> 3, k = (s & 7) << 3;
            async_b128(&As[buf][row * LDSK + k], &A[(size_t)(m0 + row) * K + k0 + k]);
        }
#pragma unroll
        for (int i = 0; i < 4; i++) {
            int s = tid + i * 256;
            int n = s >> 3, k = (s & 7) << 3;
            async_b128(&Bs[buf][n * LDSK + k], &WT[(size_t)(n0 + n) * K + k0 + k]);
        }
    };

    const int nk = K / BKT;
    stage(0, 0);
    for (int t = 0; t < nk; t++) {
        const int cur = t & 1;
        if (t + 1 < nk) { stage((t + 1) * BKT, cur ^ 1); wait_async8(); }
        else            { wait_async0(); }
        __syncthreads();

        const unsigned short* Ab = As[cur];
        const unsigned short* Bb = Bs[cur];
#pragma unroll
        for (int ks = 0; ks < 2; ks++) {       // two 32-deep WMMA slices per K-tile
            const int kb = ks * 32;
            v16bf afrag[4], bfrag[2];
#pragma unroll
            for (int mt = 0; mt < 4; mt++) {
                int row = waveRow * 64 + mt * 16 + l16;
                union { v16bf v; uint4 q[2]; } u;
                u.q[0] = *(const uint4*)&Ab[row * LDSK + kb + halfId * 8];
                u.q[1] = *(const uint4*)&Ab[row * LDSK + kb + 16 + halfId * 8];
                afrag[mt] = u.v;
            }
#pragma unroll
            for (int nt = 0; nt < 2; nt++) {
                int coln = waveCol * 32 + nt * 16 + l16;
                union { v16bf v; uint4 q[2]; } u;
                u.q[0] = *(const uint4*)&Bb[coln * LDSK + kb + halfId * 8];
                u.q[1] = *(const uint4*)&Bb[coln * LDSK + kb + 16 + halfId * 8];
                bfrag[nt] = u.v;
            }
#pragma unroll
            for (int mt = 0; mt < 4; mt++)
#pragma unroll
                for (int nt = 0; nt < 2; nt++)
                    acc[mt][nt] = __builtin_amdgcn_wmma_f32_16x16x32_bf16(
                        false, afrag[mt], false, bfrag[nt], (short)0, acc[mt][nt], false, false);
        }
        __syncthreads();
    }

    // epilogue (C/D: VGPR r -> M = r | r+8 by lane half, N = lane%16)
#pragma unroll
    for (int mt = 0; mt < 4; mt++) {
#pragma unroll
        for (int nt = 0; nt < 2; nt++) {
            int gn = n0 + waveCol * 32 + nt * 16 + l16;
            float bv = 0.0f;
            if (MODE != 2) bv = bias ? bias[gn] : 0.0f;
#pragma unroll
            for (int r = 0; r < 8; r++) {
                int gm = m0 + waveRow * 64 + mt * 16 + ((lane < 16) ? r : (r + 8));
                float v = acc[mt][nt][r] + bv;
                size_t o = (size_t)gm * N + gn;
                if (MODE == 1) v = fmaxf(v, 0.0f);
                if (MODE == 2) { if (gn < SDIM) v = fast_sigmoid(v + bias[gn]); }
                if (MODE == 3)      Cf[o] = gates[(gm >> 11) * 3 + lvl] * v;
                else if (MODE == 4) Cf[o] += gates[(gm >> 11) * 3 + lvl] * v;
                else {
                    if (OUT == 0 || OUT == 2) Cf[o] = v;
                    if (OUT == 1 || OUT == 2) Ch[o] = f32_to_bf16(v);
                }
            }
        }
    }
}

// ------------------------------------------------ sequential scan -> bf16 states
__global__ void k_scan(const float* __restrict__ A_log, const float* __restrict__ gateBx,
                       unsigned short* __restrict__ states)
{
    int b = blockIdx.x;
    int s = threadIdx.x;                       // 128 state dims
    float a = __expf(A_log[s]);
    a = fminf(fmaxf(a, 0.5f), 0.99f);
    float st = 0.0f;
    for (int t = 0; t < TLEN; t++) {
        size_t row = (size_t)(b * TLEN + t);
        float g  = gateBx[row * 256 + s];
        float bx = gateBx[row * 256 + SDIM + s];
        st = a * st + g * bx;
        st = fminf(fmaxf(st, -10.0f), 10.0f);
        states[row * SDIM + s] = f32_to_bf16(st);
    }
}

// ------------------------------------------------ layernorm of (a + (D+1)*b); OUT: 0 f32, 1 bf16
template<int OUT>
__global__ __launch_bounds__(256) void k_layernorm(
    const float* __restrict__ a, const float* __restrict__ b,
    const float* __restrict__ Dv, const float* __restrict__ g,
    const float* __restrict__ beta, float* __restrict__ outf, unsigned short* __restrict__ outh)
{
    __shared__ float red[256];
    __shared__ float red2[256];
    int m = blockIdx.x;
    int tid = threadIdx.x;
    float v[4];
    float s = 0.f, sq = 0.f;
#pragma unroll
    for (int j = 0; j < 4; j++) {
        int h = tid + j * 256;
        float scale = Dv ? (Dv[h] + 1.0f) : 1.0f;
        float val = a[(size_t)m * HDIM + h] + scale * b[(size_t)m * HDIM + h];
        v[j] = val; s += val; sq += val * val;
    }
    red[tid] = s; red2[tid] = sq;
    __syncthreads();
    for (int off = 128; off > 0; off >>= 1) {
        if (tid < off) { red[tid] += red[tid + off]; red2[tid] += red2[tid + off]; }
        __syncthreads();
    }
    float mean = red[0] * (1.0f / HDIM);
    float var  = red2[0] * (1.0f / HDIM) - mean * mean;
    float rstd = rsqrtf(var + 1e-5f);
#pragma unroll
    for (int j = 0; j < 4; j++) {
        int h = tid + j * 256;
        float o = (v[j] - mean) * rstd * g[h] + beta[h];
        if (OUT == 0) outf[(size_t)m * HDIM + h] = o;
        else          outh[(size_t)m * HDIM + h] = f32_to_bf16(o);
    }
}

// ------------------------------------------------ x mean over T
__global__ void k_xmean(const float* __restrict__ x, float* __restrict__ xmean) {
    int gid = blockIdx.x * blockDim.x + threadIdx.x;   // 4096
    if (gid >= BATCH * HDIM) return;
    int b = gid / HDIM, h = gid % HDIM;
    float s = 0.f;
    for (int t = 0; t < TLEN; t++) s += x[((size_t)b * TLEN + t) * HDIM + h];
    xmean[gid] = s * (1.0f / TLEN);
}

// ------------------------------------------------ gates = softmax(xmean @ gW + gb)
__global__ void k_gates(const float* __restrict__ xmean, const float* __restrict__ gw,
                        const float* __restrict__ gb, float* __restrict__ gates,
                        float* __restrict__ out_tail)
{
    __shared__ float part[12][32];
    __shared__ float logits[12];
    int tid = threadIdx.x;                 // 384 threads: 12 groups of 32
    int grp = tid >> 5, lane = tid & 31;
    int b = grp / 3, l = grp % 3;
    float s = 0.f;
    for (int k = lane; k < HDIM; k += 32) s += xmean[b * HDIM + k] * gw[k * 3 + l];
    part[grp][lane] = s;
    __syncthreads();
    if (tid < 12) {
        float t = 0.f;
        for (int i = 0; i < 32; i++) t += part[tid][i];
        logits[tid] = t + gb[tid % 3];
    }
    __syncthreads();
    if (tid < 4) {
        float m = fmaxf(logits[tid * 3], fmaxf(logits[tid * 3 + 1], logits[tid * 3 + 2]));
        float e0 = __expf(logits[tid * 3] - m);
        float e1 = __expf(logits[tid * 3 + 1] - m);
        float e2 = __expf(logits[tid * 3 + 2] - m);
        float inv = __builtin_amdgcn_rcpf(e0 + e1 + e2);
        gates[tid * 3 + 0] = e0 * inv; gates[tid * 3 + 1] = e1 * inv; gates[tid * 3 + 2] = e2 * inv;
        out_tail[tid * 3 + 0] = e0 * inv; out_tail[tid * 3 + 1] = e1 * inv; out_tail[tid * 3 + 2] = e2 * inv;
    }
}

// ------------------------------------------------ host
extern "C" void kernel_launch(void* const* d_in, const int* in_sizes, int n_in,
                              void* d_out, int out_size, void* d_ws, size_t ws_size,
                              hipStream_t stream)
{
    (void)in_sizes; (void)n_in; (void)out_size; (void)ws_size;
    const int CD[3] = {512, 256, 128};

    const float* x = (const float*)d_in[0];
    const float *W1[3], *b1[3], *W2[3], *b2[3];
    for (int l = 0; l < 3; l++) {
        W1[l] = (const float*)d_in[1 + l * 4]; b1[l] = (const float*)d_in[2 + l * 4];
        W2[l] = (const float*)d_in[3 + l * 4]; b2[l] = (const float*)d_in[4 + l * 4];
    }
    const float *Wd[3], *bd[3];
    for (int l = 0; l < 3; l++) { Wd[l] = (const float*)d_in[13 + l * 2]; bd[l] = (const float*)d_in[14 + l * 2]; }
    const float *Alog[3], *Bm[3], *Cm[3], *Dm[3], *gW[3], *gb[3], *oW[3], *ob[3], *lng[3], *lnb[3];
    for (int l = 0; l < 3; l++) {
        int base = 19 + l * 10;
        Alog[l] = (const float*)d_in[base + 0]; Bm[l] = (const float*)d_in[base + 1];
        Cm[l]   = (const float*)d_in[base + 2]; Dm[l] = (const float*)d_in[base + 3];
        gW[l]   = (const float*)d_in[base + 4]; gb[l] = (const float*)d_in[base + 5];
        oW[l]   = (const float*)d_in[base + 6]; ob[l] = (const float*)d_in[base + 7];
        lng[l]  = (const float*)d_in[base + 8]; lnb[l] = (const float*)d_in[base + 9];
    }
    const float* gateW = (const float*)d_in[49];
    const float* gateb = (const float*)d_in[50];
    const float* normg = (const float*)d_in[51];
    const float* normb = (const float*)d_in[52];

    // workspace layout
    char* p = (char*)d_ws;
    auto alloc_h = [&](size_t n) -> unsigned short* { unsigned short* r = (unsigned short*)p; p += ((n * 2 + 255) / 256) * 256; return r; };
    auto alloc_f = [&](size_t n) -> float*          { float* r = (float*)p;          p += ((n * 4 + 255) / 256) * 256; return r; };
    unsigned short *wW1[3], *wW2[3], *wWd[3], *wgB[3], *wC[3], *woW[3];
    for (int l = 0; l < 3; l++) {
        size_t c = (size_t)CD[l];
        wW1[l] = alloc_h(1024 * c); wW2[l] = alloc_h(c * c); wWd[l] = alloc_h(c * 1024);
        wgB[l] = alloc_h((size_t)256 * 1024);      // [gW^T ; B^T] : 256 rows x 1024
        wC[l]  = alloc_h((size_t)1024 * 128);      // C^T : 1024 rows x 128
        woW[l] = alloc_h((size_t)1024 * 1024);
    }
    unsigned short* x_bf    = alloc_h((size_t)MROWS * 1024);
    unsigned short* buf_a   = alloc_h((size_t)MROWS * 512);
    unsigned short* buf_b   = alloc_h((size_t)MROWS * 512);
    unsigned short* dec_bf  = alloc_h((size_t)MROWS * 1024);
    unsigned short* st_bf   = alloc_h((size_t)MROWS * 128);
    unsigned short* ln_bf   = alloc_h((size_t)MROWS * 1024);
    float* dec     = alloc_f((size_t)MROWS * 1024);
    float* gateBx  = alloc_f((size_t)MROWS * 256);
    float* ybuf    = alloc_f((size_t)MROWS * 1024);
    float* xmean   = alloc_f(4096);
    float* gates   = alloc_f(16);

    auto convT = [&](const float* src, unsigned short* dst, int K, int N) {
        size_t n = (size_t)K * N;
        int blocks = (int)((n + 255) / 256); if (blocks > 2048) blocks = 2048;
        k_convT<<<dim3(blocks), dim3(256), 0, stream>>>(src, dst, K, N);
    };
    for (int l = 0; l < 3; l++) {
        int c = CD[l];
        convT(W1[l], wW1[l], 1024, c);
        convT(W2[l], wW2[l], c, c);
        convT(Wd[l], wWd[l], c, 1024);
        convT(gW[l], wgB[l],              1024, 128);   // rows 0..127  = gW^T
        convT(Bm[l], wgB[l] + 128 * 1024, 1024, 128);   // rows 128..255 = B^T
        convT(Cm[l], wC[l], 128, 1024);
        convT(oW[l], woW[l], 1024, 1024);
    }
    k_conv<<<dim3(2048), dim3(256), 0, stream>>>(x, x_bf, MROWS * HDIM);

    float* out_main = (float*)d_out;
    float* out_tail = out_main + (size_t)MROWS * HDIM;
    k_xmean<<<dim3(16), dim3(256), 0, stream>>>(x, xmean);
    k_gates<<<dim3(1), dim3(384), 0, stream>>>(xmean, gateW, gateb, gates, out_tail);

    auto grid = [](int M, int N) { return dim3(N / BN, M / BM); };

    for (int l = 0; l < 3; l++) {
        int c = CD[l];
        k_gemm<1, 1><<<grid(MROWS, c), 256, 0, stream>>>(x_bf, wW1[l], b1[l], nullptr, buf_a, MROWS, c, 1024, nullptr, 0);
        k_gemm<0, 1><<<grid(MROWS, c), 256, 0, stream>>>(buf_a, wW2[l], b2[l], nullptr, buf_b, MROWS, c, c, nullptr, 0);
        k_gemm<1, 2><<<grid(MROWS, 1024), 256, 0, stream>>>(buf_b, wWd[l], bd[l], dec, dec_bf, MROWS, 1024, c, nullptr, 0);
        k_gemm<2, 0><<<grid(MROWS, 256), 256, 0, stream>>>(dec_bf, wgB[l], gb[l], gateBx, nullptr, MROWS, 256, 1024, nullptr, 0);
        k_scan<<<dim3(BATCH), dim3(SDIM), 0, stream>>>(Alog[l], gateBx, st_bf);
        k_gemm<0, 0><<<grid(MROWS, 1024), 256, 0, stream>>>(st_bf, wC[l], nullptr, ybuf, nullptr, MROWS, 1024, 128, nullptr, 0);
        k_layernorm<1><<<dim3(MROWS), dim3(256), 0, stream>>>(ybuf, dec, Dm[l], lng[l], lnb[l], nullptr, ln_bf);
        if (l == 0)
            k_gemm<3, 0><<<grid(MROWS, 1024), 256, 0, stream>>>(ln_bf, woW[l], ob[l], out_main, nullptr, MROWS, 1024, 1024, gates, l);
        else
            k_gemm<4, 0><<<grid(MROWS, 1024), 256, 0, stream>>>(ln_bf, woW[l], ob[l], out_main, nullptr, MROWS, 1024, 1024, gates, l);
    }
    k_layernorm<0><<<dim3(MROWS), dim3(256), 0, stream>>>(out_main, x, nullptr, normg, normb, out_main, nullptr);
}
